// Branch_74560632258692
// MI455X (gfx1250) — matrix operand order repaired and checked by reading, not compiled, not run on hardware
//
// ============================================================================
// Point Cloud Transformer forward for MI455X (gfx1250, CDNA5, wave32).
//
// All channel-matmuls lower to V_WMMA_F32_16X16X32_F16 (every K is a multiple
// of 32, every row/col count a multiple of 32). Each wave computes a 32x32
// output patch (2x2 register tiling -> 4 WMMAs per K-step, 2x operand reuse).
// GEMM variants are compile-time templates so the inner loop is branch-free:
// float4 loads + v_cvt_pk_f16_f32 packing + 4x wmma. Activations kept in
// (B, N, C) row-major; SA-layer energy = q @ q^T (shared wqk weights);
// x_r = attn^T @ v via the transposed-A path.
// ============================================================================
#include <hip/hip_runtime.h>
#include <hip/hip_bf16.h>

typedef __attribute__((ext_vector_type(16))) _Float16 v16h;
typedef __attribute__((ext_vector_type(8)))  float    v8f;

__device__ __forceinline__ void pack4(v16h& v, int base, const float4 f)
{
    v[base + 0] = (_Float16)f.x;
    v[base + 1] = (_Float16)f.y;
    v[base + 2] = (_Float16)f.z;
    v[base + 3] = (_Float16)f.w;
}

// A fragment, row-major A: ISA 16-bit A 16x32 layout: K = e + (e&8) + 8*hi.
// Lane's 16 halves are two contiguous 8-float runs -> 4x float4 loads.
template <int HS>
__device__ __forceinline__ v16h ldA(const float* __restrict__ A, int lda,
                                    const float* __restrict__ Sub, int ldsub,
                                    int row, int k0, int hi)
{
    const float4* A4 = (const float4*)(A + (long long)row * lda + k0);
    float4 a0 = A4[hi * 2 + 0];
    float4 a1 = A4[hi * 2 + 1];
    float4 a2 = A4[hi * 2 + 4];
    float4 a3 = A4[hi * 2 + 5];
    if constexpr (HS) {
        const float4* S4 = (const float4*)(Sub + (long long)row * ldsub + k0);
        const float4 s0 = S4[hi * 2 + 0];
        const float4 s1 = S4[hi * 2 + 1];
        const float4 s2 = S4[hi * 2 + 4];
        const float4 s3 = S4[hi * 2 + 5];
        a0.x -= s0.x; a0.y -= s0.y; a0.z -= s0.z; a0.w -= s0.w;
        a1.x -= s1.x; a1.y -= s1.y; a1.z -= s1.z; a1.w -= s1.w;
        a2.x -= s2.x; a2.y -= s2.y; a2.z -= s2.z; a2.w -= s2.w;
        a3.x -= s3.x; a3.y -= s3.y; a3.z -= s3.z; a3.w -= s3.w;
    }
    v16h v;
    pack4(v, 0, a0); pack4(v, 4, a1); pack4(v, 8, a2); pack4(v, 12, a3);
    return v;
}

// A fragment, transposed A (attn^T): strided scalar gathers.
__device__ __forceinline__ v16h ldAt(const float* __restrict__ A, int lda,
                                     int row, int k0, int hi)
{
    v16h v;
#pragma unroll
    for (int e = 0; e < 16; ++e) {
        const int ka = k0 + e + (e & 8) + (hi << 3);
        v[e] = (_Float16)A[(long long)ka * lda + row];
    }
    return v;
}

// B fragment, weight-style (O x K): same K order as A: K = e + (e&8) + 8*hi.
// Two contiguous 8-float runs -> 4x float4 loads.
__device__ __forceinline__ v16h ldBw(const float* __restrict__ Bm, int ldb,
                                     int col, int k0, int hi)
{
    const float4* B4 = (const float4*)(Bm + (long long)col * ldb + k0);
    v16h v;
    pack4(v, 0,  B4[hi * 2 + 0]);
    pack4(v, 4,  B4[hi * 2 + 1]);
    pack4(v, 8,  B4[hi * 2 + 4]);
    pack4(v, 12, B4[hi * 2 + 5]);
    return v;
}

// B fragment, K-major (K x O): strided scalar gathers.
__device__ __forceinline__ v16h ldBn(const float* __restrict__ Bm, int ldb,
                                     int col, int k0, int hi)
{
    v16h v;
#pragma unroll
    for (int e = 0; e < 16; ++e) {
        const int kb = k0 + e + (e & 8) + (hi << 3);
        v[e] = (_Float16)Bm[(long long)kb * ldb + col];
    }
    return v;
}

// ---------------------------------------------------------------------------
// One wave computes one 32x32 patch (2x2 tiles) of Out = epilogue(A @ B).
//   A[r,k] = (AT ? Ap[k*lda+r] : Ap[r*lda+k]) - (HS ? Sub[r*ldsub+k] : 0)
//   B[k,o] = (BT ? Bp[o*ldb+k] : Bp[k*ldb+o])      (BT = weight-style O x K)
//   epilogue: y=acc; +bias[o]; y=scale[o]*y+shift[o]; ACT(1:relu,2:leaky0.2);
//             y += Res[r*ldr+o];  Out[r*ldo + colOff + o] = y
// grid = (O/32, R/32, batch); block = 32 (single wave -> EXEC all ones).
// ---------------------------------------------------------------------------
template <int AT, int HS, int BT, int ACT, int HB, int HSC, int HR>
__global__ void __launch_bounds__(32)
wmma_gemm(const float* __restrict__ Ap, long long sA, int lda,
          const float* __restrict__ Sub, int ldsub,
          const float* __restrict__ Bp, long long sB, int ldb,
          const float* __restrict__ bias, const float* __restrict__ scale,
          const float* __restrict__ shift,
          const float* __restrict__ Res, int ldr,
          float* __restrict__ Out, long long sOut, int ldo, int colOff,
          int K)
{
    const int oT = blockIdx.x, rT = blockIdx.y, b = blockIdx.z;
    const float* A  = Ap + (long long)b * sA;
    const float* Bm = Bp + (long long)b * sB;

    const int lane = threadIdx.x;
    const int hi   = lane >> 4;        // lane half: 0 or 1
    const int l16  = lane & 15;
    const int row0 = rT * 32 + l16;    // A rows held by this lane
    const int row1 = row0 + 16;
    const int col0 = oT * 32 + l16;    // output columns held by this lane
    const int col1 = col0 + 16;

    v8f acc[2][2] = {{{}, {}}, {{}, {}}};   // [rowTile][colTile]
    for (int k0 = 0; k0 < K; k0 += 32) {
        v16h a0, a1, b0, b1;
        if constexpr (!AT) {
            a0 = ldA<HS>(A, lda, Sub, ldsub, row0, k0, hi);
            a1 = ldA<HS>(A, lda, Sub, ldsub, row1, k0, hi);
        } else {
            a0 = ldAt(A, lda, row0, k0, hi);
            a1 = ldAt(A, lda, row1, k0, hi);
        }
        if constexpr (BT) {
            b0 = ldBw(Bm, ldb, col0, k0, hi);
            b1 = ldBw(Bm, ldb, col1, k0, hi);
        } else {
            b0 = ldBn(Bm, ldb, col0, k0, hi);
            b1 = ldBn(Bm, ldb, col1, k0, hi);
        }
        acc[0][0] = __builtin_amdgcn_wmma_f32_16x16x32_f16(
                        false, a0, false, b0, (short)0, acc[0][0], false, false);
        acc[0][1] = __builtin_amdgcn_wmma_f32_16x16x32_f16(
                        false, a0, false, b1, (short)0, acc[0][1], false, false);
        acc[1][0] = __builtin_amdgcn_wmma_f32_16x16x32_f16(
                        false, a1, false, b0, (short)0, acc[1][0], false, false);
        acc[1][1] = __builtin_amdgcn_wmma_f32_16x16x32_f16(
                        false, a1, false, b1, (short)0, acc[1][1], false, false);
    }

#pragma unroll
    for (int ci = 0; ci < 2; ++ci) {
        const int col = ci ? col1 : col0;
        float bi = 0.f, sc = 1.f, sh = 0.f;
        if constexpr (HB)  bi = bias[col];
        if constexpr (HSC) { sc = scale[col]; sh = shift[col]; }
#pragma unroll
        for (int ri = 0; ri < 2; ++ri) {
#pragma unroll
            for (int r = 0; r < 8; ++r) {
                const int row = rT * 32 + ri * 16 + r + (hi << 3);  // ISA C/D layout
                float y = acc[ri][ci][r];
                if constexpr (HB)  y += bi;
                if constexpr (HSC) y = sc * y + sh;
                if constexpr (ACT == 1)      y = fmaxf(y, 0.f);
                else if constexpr (ACT == 2) y = y > 0.f ? y : 0.2f * y;
                if constexpr (HR)  y += Res[(long long)row * ldr + col];
                Out[(long long)b * sOut + (long long)row * ldo + colOff + col] = y;
            }
        }
    }
}

template <int AT, int HS, int BT, int ACT, int HB, int HSC, int HR>
static inline void gemm(hipStream_t st,
                        const float* A, long long sA, int lda,
                        const float* Sub, int ldsub,
                        const float* B, long long sB, int ldb,
                        const float* bias, const float* scale, const float* shift,
                        const float* Res, int ldr,
                        float* Out, long long sOut, int ldo, int colOff,
                        int R, int K, int O, int batch)
{
    dim3 g(O / 32, R / 32, batch);
    wmma_gemm<AT, HS, BT, ACT, HB, HSC, HR><<<g, 32, 0, st>>>(
        A, sA, lda, Sub, ldsub, B, sB, ldb, bias, scale, shift,
        Res, ldr, Out, sOut, ldo, colOff, K);
}

// --------------------------- conv1 (K=7, scalar) ---------------------------
__global__ void conv1_kernel(const float* __restrict__ x,   // (B*N, 7)
                             const float* __restrict__ w,   // (64, 7)
                             const float* __restrict__ s,
                             const float* __restrict__ bb,
                             float* __restrict__ out,       // (B*N, 64)
                             float* __restrict__ xyz,       // (B*N, 3)
                             long long total)               // B*N*64
{
    long long i = (long long)blockIdx.x * blockDim.x + threadIdx.x;
    if (i >= total) return;
    long long bn = i >> 6;
    int o = (int)(i & 63);
    const float* xr = x + bn * 7;
    float acc = 0.f;
#pragma unroll
    for (int c = 0; c < 7; ++c) acc += w[o * 7 + c] * xr[c];
    acc = s[o] * acc + bb[o];
    out[i] = acc > 0.f ? acc : 0.f;
    if (o < 3) xyz[bn * 3 + o] = xr[o];
}

// ------------------------- Farthest point sampling -------------------------
#define FPS_T 256
__global__ void fps_kernel(const float* __restrict__ xyz, int N, int npoint,
                           int* __restrict__ outIdx)
{
    __shared__ float dist[4096];
    __shared__ float rv[FPS_T];
    __shared__ int   ri[FPS_T];
    __shared__ float cx, cy, cz;
    __shared__ int   far;
    const int b = blockIdx.x, t = threadIdx.x;
    const float* P = xyz + (long long)b * N * 3;
    for (int i = t; i < N; i += FPS_T) dist[i] = 1e10f;
    if (t == 0) far = 0;
    __syncthreads();
    for (int it = 0; it < npoint; ++it) {
        if (t == 0) {
            outIdx[b * npoint + it] = far;          // record incoming 'far'
            cx = P[far * 3 + 0]; cy = P[far * 3 + 1]; cz = P[far * 3 + 2];
        }
        __syncthreads();
        float bv = -1.f; int bi = 0;
        for (int i = t; i < N; i += FPS_T) {
            const float dx = P[i * 3 + 0] - cx;
            const float dy = P[i * 3 + 1] - cy;
            const float dz = P[i * 3 + 2] - cz;
            const float nd = fminf(dist[i], dx * dx + dy * dy + dz * dz);
            dist[i] = nd;
            if (nd > bv) { bv = nd; bi = i; }
        }
        rv[t] = bv; ri[t] = bi;
        __syncthreads();
        for (int sft = FPS_T / 2; sft > 0; sft >>= 1) {
            if (t < sft && rv[t + sft] > rv[t]) { rv[t] = rv[t + sft]; ri[t] = ri[t + sft]; }
            __syncthreads();
        }
        if (t == 0) far = ri[0];
        __syncthreads();
    }
}

__global__ void gather_xyz_kernel(const float* __restrict__ xyz,
                                  const int* __restrict__ idx,
                                  float* __restrict__ out, int N, int S, int total)
{
    int i = blockIdx.x * blockDim.x + threadIdx.x;
    if (i >= total) return;
    int b = i / S;
    int g = idx[i];
    const float* p = xyz + ((long long)b * N + g) * 3;
    float* o = out + (long long)i * 3;
    o[0] = p[0]; o[1] = p[1]; o[2] = p[2];
}

// --------------------------- kNN (32 smallest d2) ---------------------------
#define KNN_T 128
__global__ void knn_kernel(const float* __restrict__ xyz,   // (B, N, 3)
                           const float* __restrict__ cent,  // (B, S, 3)
                           int N, int S, int* __restrict__ knn) // (B, S, 32)
{
    __shared__ float d2[4096];
    __shared__ float rv[KNN_T];
    __shared__ int   ri[KNN_T];
    const int b = blockIdx.y, s = blockIdx.x, t = threadIdx.x;
    const float* P = xyz + (long long)b * N * 3;
    const float* C = cent + ((long long)b * S + s) * 3;
    const float cx = C[0], cy = C[1], cz = C[2];
    for (int i = t; i < N; i += KNN_T) {
        const float dx = P[i * 3 + 0] - cx;
        const float dy = P[i * 3 + 1] - cy;
        const float dz = P[i * 3 + 2] - cz;
        d2[i] = dx * dx + dy * dy + dz * dz;
    }
    __syncthreads();
    for (int kk = 0; kk < 32; ++kk) {
        float bv = 3.4e38f; int bi = 0;
        for (int i = t; i < N; i += KNN_T)
            if (d2[i] < bv) { bv = d2[i]; bi = i; }
        rv[t] = bv; ri[t] = bi;
        __syncthreads();
        for (int sft = KNN_T / 2; sft > 0; sft >>= 1) {
            if (t < sft && rv[t + sft] < rv[t]) { rv[t] = rv[t + sft]; ri[t] = ri[t + sft]; }
            __syncthreads();
        }
        if (t == 0) {
            knn[((long long)b * S + s) * 32 + kk] = ri[0];
            d2[ri[0]] = 3.4e38f;
        }
        __syncthreads();
    }
}

// ----------------- group: feat = [gathered - center, center] ----------------
__global__ void group_feat_kernel(const float* __restrict__ points, // (B,N,C)
                                  const int* __restrict__ fpsIdx,   // (B,S)
                                  const int* __restrict__ knn,      // (B,S,32)
                                  float* __restrict__ feat,         // (B,S,32,2C)
                                  int N, int S, int C)
{
    __shared__ float np[256];
    const int b = blockIdx.y, s = blockIdx.x, t = threadIdx.x;
    const int ci = fpsIdx[b * S + s];
    const float* Pb = points + (long long)b * N * C;
    for (int c = t; c < C; c += blockDim.x) np[c] = Pb[(long long)ci * C + c];
    __syncthreads();
    const int* kn = knn + ((long long)b * S + s) * 32;
    float* F = feat + ((long long)b * S + s) * 32 * 2 * C;
    for (int k = 0; k < 32; ++k) {
        const float* G = Pb + (long long)kn[k] * C;
        for (int c = t; c < C; c += blockDim.x) {
            const float n = np[c];
            F[(long long)k * 2 * C + c]     = G[c] - n;
            F[(long long)k * 2 * C + C + c] = n;
        }
    }
}

// ------------------------------ max over k=32 -------------------------------
__global__ void maxpool_k_kernel(const float* __restrict__ g,  // (BS*32, O)
                                 float* __restrict__ f,        // (BS, O)
                                 int O)
{
    const long long bs = blockIdx.x;
    const int o = blockIdx.y * blockDim.x + threadIdx.x;
    if (o >= O) return;
    const float* G = g + bs * 32 * O + o;
    float m = G[0];
    for (int k = 1; k < 32; ++k) m = fmaxf(m, G[(long long)k * O]);
    f[bs * O + o] = m;
}

// ------------------------- softmax over rows of 256 -------------------------
__global__ void softmax_row_kernel(float* __restrict__ e)
{
    __shared__ float red[256];
    const int t = threadIdx.x;
    float* row = e + (long long)blockIdx.x * 256;
    const float v = row[t];
    red[t] = v; __syncthreads();
    for (int s = 128; s > 0; s >>= 1) { if (t < s) red[t] = fmaxf(red[t], red[t + s]); __syncthreads(); }
    const float mx = red[0]; __syncthreads();
    const float ex = __expf(v - mx);
    red[t] = ex; __syncthreads();
    for (int s = 128; s > 0; s >>= 1) { if (t < s) red[t] += red[t + s]; __syncthreads(); }
    row[t] = ex / red[0];
}

__global__ void attn_colsum_kernel(const float* __restrict__ attn,
                                   float* __restrict__ cs, int Nn)
{
    const int b = blockIdx.x, m = threadIdx.x;
    const float* A = attn + (long long)b * Nn * 256;
    float s = 0.f;
    for (int n = 0; n < Nn; ++n) s += A[(long long)n * 256 + m];
    cs[b * 256 + m] = s;
}

__global__ void attn_colnorm_kernel(float* __restrict__ attn,
                                    const float* __restrict__ cs, int Nn)
{
    const long long row = blockIdx.x;
    const int b = (int)(row / Nn);
    attn[row * 256 + threadIdx.x] /= (1e-9f + cs[b * 256 + threadIdx.x]);
}

// ------------------------------- misc copies --------------------------------
__global__ void copy_cols_kernel(const float* __restrict__ src, int lsrc,
                                 float* __restrict__ dst, int ldst, int off)
{
    dst[(long long)blockIdx.x * ldst + off + threadIdx.x] =
        src[(long long)blockIdx.x * lsrc + threadIdx.x];
}

__global__ void max_over_n_kernel(const float* __restrict__ h, // (B*Nn, O)
                                  float* __restrict__ out,     // (B, O)
                                  int Nn, int O)
{
    const int b = blockIdx.y;
    const int o = blockIdx.x * blockDim.x + threadIdx.x;
    if (o >= O) return;
    const float* H = h + (long long)b * Nn * O + o;
    float m = H[0];
    for (int n = 1; n < Nn; ++n) m = fmaxf(m, H[(long long)n * O]);
    out[(long long)b * O + o] = m;
}

// ---------------------------------------------------------------------------
extern "C" void kernel_launch(void* const* d_in, const int* in_sizes, int n_in,
                              void* d_out, int out_size, void* d_ws, size_t ws_size,
                              hipStream_t stream)
{
    (void)in_sizes; (void)n_in; (void)out_size; (void)ws_size;
    // Input order: depth-first insertion order of setup_inputs() dict.
    int ii = 0;
    const float* X       = (const float*)d_in[ii++]; // (8,4096,7)
    const float* conv1_w = (const float*)d_in[ii++];
    const float* bn1_s   = (const float*)d_in[ii++];
    const float* bn1_b   = (const float*)d_in[ii++];
    const float* conv2_w = (const float*)d_in[ii++];
    const float* bn2_s   = (const float*)d_in[ii++];
    const float* bn2_b   = (const float*)d_in[ii++];
    const float* l0w1 = (const float*)d_in[ii++]; const float* l0s1 = (const float*)d_in[ii++];
    const float* l0b1 = (const float*)d_in[ii++]; const float* l0w2 = (const float*)d_in[ii++];
    const float* l0s2 = (const float*)d_in[ii++]; const float* l0b2 = (const float*)d_in[ii++];
    const float* l1w1 = (const float*)d_in[ii++]; const float* l1s1 = (const float*)d_in[ii++];
    const float* l1b1 = (const float*)d_in[ii++]; const float* l1w2 = (const float*)d_in[ii++];
    const float* l1s2 = (const float*)d_in[ii++]; const float* l1b2 = (const float*)d_in[ii++];
    const float* pt1w = (const float*)d_in[ii++]; const float* pt1s = (const float*)d_in[ii++];
    const float* pt1b = (const float*)d_in[ii++]; const float* pt2w = (const float*)d_in[ii++];
    const float* pt2s = (const float*)d_in[ii++]; const float* pt2b = (const float*)d_in[ii++];
    struct SaP { const float *wqk, *wv, *bv, *wt, *bt, *s, *b; } sa[4];
    for (int l = 0; l < 4; ++l) {
        sa[l].wqk = (const float*)d_in[ii++]; sa[l].wv = (const float*)d_in[ii++];
        sa[l].bv  = (const float*)d_in[ii++]; sa[l].wt = (const float*)d_in[ii++];
        sa[l].bt  = (const float*)d_in[ii++]; sa[l].s  = (const float*)d_in[ii++];
        sa[l].b   = (const float*)d_in[ii++];
    }
    const float* cf_w = (const float*)d_in[ii++];
    const float* cf_s = (const float*)d_in[ii++];
    const float* cf_b = (const float*)d_in[ii++];

    const int B = 8, N = 4096, S0 = 512, S1 = 256;

    // Bump allocator over d_ws (peak ~321 MB).
    char* wp = (char*)d_ws;
    auto falloc = [&](size_t elems) -> float* {
        float* r = (float*)wp; wp += ((elems * 4 + 255) & ~(size_t)255); return r;
    };
    auto ialloc = [&](size_t elems) -> int* {
        int* r = (int*)wp; wp += ((elems * 4 + 255) & ~(size_t)255); return r;
    };

    float* xyz0   = falloc((size_t)B * N * 3);
    float* h1     = falloc((size_t)B * N * 64);
    float* pts    = falloc((size_t)B * N * 64);
    int*   fidx0  = ialloc((size_t)B * S0);
    float* nxyz0  = falloc((size_t)B * S0 * 3);
    int*   knn0   = ialloc((size_t)B * S0 * 32);
    float* bufA   = falloc((size_t)B * S0 * 32 * 256);   // feat / g2 ping-pong
    float* bufB   = falloc((size_t)B * S0 * 32 * 256);   // g1
    float* f0     = falloc((size_t)B * S0 * 128);
    int*   fidx1  = ialloc((size_t)B * S1);
    float* nxyz1  = falloc((size_t)B * S1 * 3);
    int*   knn1   = ialloc((size_t)B * S1 * 32);
    float* f1     = falloc((size_t)B * S1 * 256);
    float* hA     = falloc((size_t)B * S1 * 256);
    float* hB     = falloc((size_t)B * S1 * 256);
    float* qb     = falloc((size_t)B * S1 * 64);
    float* attn   = falloc((size_t)B * S1 * 256);
    float* csum   = falloc((size_t)B * 256);
    float* vb     = falloc((size_t)B * S1 * 256);
    float* xr     = falloc((size_t)B * S1 * 256);
    float* concat = falloc((size_t)B * S1 * 1280);
    float* hcf    = falloc((size_t)B * S1 * 1024);

    // 1) conv1 + bn + relu (K=7, scalar) and xyz extraction
    {
        long long tot = (long long)B * N * 64;
        conv1_kernel<<<dim3((unsigned)((tot + 255) / 256)), 256, 0, stream>>>(
            X, conv1_w, bn1_s, bn1_b, h1, xyz0, tot);
    }
    // 2) conv2 + bn + relu -> pts (B*N x 64)
    gemm<0,0,1, 1,0,1,0>(stream, h1, 0, 64, nullptr, 0, conv2_w, 0, 64,
                         nullptr, bn2_s, bn2_b, nullptr, 0,
                         pts, 0, 64, 0, B * N, 64, 64, 1);
    // 3) FPS 512 on xyz0
    fps_kernel<<<B, FPS_T, 0, stream>>>(xyz0, N, S0, fidx0);
    gather_xyz_kernel<<<(B * S0 + 255) / 256, 256, 0, stream>>>(xyz0, fidx0, nxyz0, N, S0, B * S0);
    // 4) kNN32 and grouping -> feat (B,512,32,128) in bufA
    knn_kernel<<<dim3(S0, B), KNN_T, 0, stream>>>(xyz0, nxyz0, N, S0, knn0);
    group_feat_kernel<<<dim3(S0, B), 128, 0, stream>>>(pts, fidx0, knn0, bufA, N, S0, 64);
    // 5) local_op l0: two GEMMs (R=131072, K=O=128) + max over k
    gemm<0,0,1, 1,0,1,0>(stream, bufA, 0, 128, nullptr, 0, l0w1, 0, 128,
                         nullptr, l0s1, l0b1, nullptr, 0,
                         bufB, 0, 128, 0, B * S0 * 32, 128, 128, 1);
    gemm<0,0,1, 1,0,1,0>(stream, bufB, 0, 128, nullptr, 0, l0w2, 0, 128,
                         nullptr, l0s2, l0b2, nullptr, 0,
                         bufA, 0, 128, 0, B * S0 * 32, 128, 128, 1);
    maxpool_k_kernel<<<dim3(B * S0, 1), 256, 0, stream>>>(bufA, f0, 128);
    // 6) FPS 256 on nxyz0, kNN over 512 pts, group with f0 -> feat (B,256,32,256)
    fps_kernel<<<B, FPS_T, 0, stream>>>(nxyz0, S0, S1, fidx1);
    gather_xyz_kernel<<<(B * S1 + 255) / 256, 256, 0, stream>>>(nxyz0, fidx1, nxyz1, S0, S1, B * S1);
    knn_kernel<<<dim3(S1, B), KNN_T, 0, stream>>>(nxyz0, nxyz1, S0, S1, knn1);
    group_feat_kernel<<<dim3(S1, B), 128, 0, stream>>>(f0, fidx1, knn1, bufA, S0, S1, 128);
    // 7) local_op l1 (R=65536, K=O=256) + max over k -> f1
    gemm<0,0,1, 1,0,1,0>(stream, bufA, 0, 256, nullptr, 0, l1w1, 0, 256,
                         nullptr, l1s1, l1b1, nullptr, 0,
                         bufB, 0, 256, 0, B * S1 * 32, 256, 256, 1);
    gemm<0,0,1, 1,0,1,0>(stream, bufB, 0, 256, nullptr, 0, l1w2, 0, 256,
                         nullptr, l1s2, l1b2, nullptr, 0,
                         bufA, 0, 256, 0, B * S1 * 32, 256, 256, 1);
    maxpool_k_kernel<<<dim3(B * S1, 1), 256, 0, stream>>>(bufA, f1, 256);
    // 8) pt convs -> SA input hB
    gemm<0,0,1, 1,0,1,0>(stream, f1, 0, 256, nullptr, 0, pt1w, 0, 256,
                         nullptr, pt1s, pt1b, nullptr, 0,
                         hA, 0, 256, 0, B * S1, 256, 256, 1);
    gemm<0,0,1, 1,0,1,0>(stream, hA, 0, 256, nullptr, 0, pt2w, 0, 256,
                         nullptr, pt2s, pt2b, nullptr, 0,
                         hB, 0, 256, 0, B * S1, 256, 256, 1);
    // 9) 4 SA layers; each writes its output into concat[:, l*256 : l*256+256]
    for (int l = 0; l < 4; ++l) {
        const float* Xl = (l == 0) ? hB : (concat + (l - 1) * 256);
        const int ldx   = (l == 0) ? 256 : 1280;
        // q = X . wqk^T (2048 x 64); k shares weights -> energy = q q^T
        gemm<0,0,1, 0,0,0,0>(stream, Xl, 0, ldx, nullptr, 0, sa[l].wqk, 0, 256,
                             nullptr, nullptr, nullptr, nullptr, 0,
                             qb, 0, 64, 0, B * S1, 256, 64, 1);
        gemm<0,0,1, 0,0,0,0>(stream, qb, (long long)S1 * 64, 64, nullptr, 0,
                             qb, (long long)S1 * 64, 64,
                             nullptr, nullptr, nullptr, nullptr, 0,
                             attn, (long long)S1 * 256, 256, 0, S1, 64, 256, B);
        softmax_row_kernel<<<B * S1, 256, 0, stream>>>(attn);
        attn_colsum_kernel<<<B, 256, 0, stream>>>(attn, csum, S1);
        attn_colnorm_kernel<<<B * S1, 256, 0, stream>>>(attn, csum, S1);
        // v = X . wv^T + bv
        gemm<0,0,1, 0,1,0,0>(stream, Xl, 0, ldx, nullptr, 0, sa[l].wv, 0, 256,
                             sa[l].bv, nullptr, nullptr, nullptr, 0,
                             vb, 0, 256, 0, B * S1, 256, 256, 1);
        // x_r = attn^T . v   (per batch)
        gemm<1,0,0, 0,0,0,0>(stream, attn, (long long)S1 * 256, 256, nullptr, 0,
                             vb, (long long)S1 * 256, 256,
                             nullptr, nullptr, nullptr, nullptr, 0,
                             xr, (long long)S1 * 256, 256, 0, S1, 256, 256, B);
        // out = X + relu(s*((X - x_r).wt^T + bt) + b)  -> concat col block l
        gemm<0,1,1, 1,1,1,1>(stream, Xl, 0, ldx, xr, 256, sa[l].wt, 0, 256,
                             sa[l].bt, sa[l].s, sa[l].b, Xl, ldx,
                             concat, 0, 1280, l * 256, B * S1, 256, 256, 1);
    }
    // 10) append f1 -> concat[:, 1024:1280]
    copy_cols_kernel<<<B * S1, 256, 0, stream>>>(f1, 256, concat, 1280, 1024);
    // 11) conv_fuse: (2048 x 1280) . (1024 x 1280)^T, scale/shift, leaky 0.2
    gemm<0,0,1, 2,0,1,0>(stream, concat, 0, 1280, nullptr, 0, cf_w, 0, 1280,
                         nullptr, cf_s, cf_b, nullptr, 0,
                         hcf, 0, 1024, 0, B * S1, 1280, 1024, 1);
    // 12) global max over n -> (8, 1024)
    max_over_n_kernel<<<dim3(1024 / 256, B), 256, 0, stream>>>(hcf, (float*)d_out, S1, 1024);
}
